// DoMINO_62732292325647
// MI455X (gfx1250) — matrix operand order repaired and checked by reading, not compiled, not run on hardware
//
#include <hip/hip_runtime.h>
#include <hip/hip_bf16.h>
#include <hip/hip_fp16.h>
#include <math.h>

typedef __attribute__((ext_vector_type(16))) _Float16 v16h;
typedef __attribute__((ext_vector_type(8)))  float    v8f;
typedef __attribute__((ext_vector_type(2)))  _Float16 h2;
typedef __attribute__((ext_vector_type(2)))  __fp16   fp16x2;

#define FRAGS_L1 16      // 8 fb x 2 kb   (48->64 padded K)
#define FRAGS_Lmid 32    // 8 fb x 4 kb
#define FRAGS_L5 4       // 1 fb x 4 kb   (4->16 padded N)
#define N_FRAGS (FRAGS_L1 + 3*FRAGS_Lmid + FRAGS_L5)   // 116
#define FRAG_HALVES 512  // 32 lanes * 16 halves
#define WS_HALVES (N_FRAGS * FRAG_HALVES)              // 59392

// ---------------------------------------------------------------------------
// Prep kernel: transpose + f16-convert + zero-pad the 5 weight matrices into
// per-lane-contiguous WMMA A-fragment order (A = W^T tile, 16 outf x 32 inf).
// A-frag layout used here (16-bit A 16x32, contiguous k, not the ISA table order):
//   lane L: row m = L&15 (out feature), K contiguous (L>>4)*16 + {0..15}.
//   half h: k = kb*32 + (L>>4)*16 + h
// ---------------------------------------------------------------------------
__global__ void domino_prep_weights(const float* __restrict__ W1,
                                    const float* __restrict__ W2,
                                    const float* __restrict__ W3,
                                    const float* __restrict__ W4,
                                    const float* __restrict__ W5,
                                    _Float16* __restrict__ ws, int total) {
    int idx = blockIdx.x * blockDim.x + threadIdx.x;
    if (idx >= total) return;
    int t    = idx >> 9;          // fragment id
    int lane = (idx >> 4) & 31;
    int h    = idx & 15;

    const float* W; int fb, kb, Kdim, Nout;
    if (t < 16)        { W = W1; int u = t;       fb = u >> 1; kb = u & 1; Kdim = 48;  Nout = 128; }
    else if (t < 48)   { W = W2; int u = t - 16;  fb = u >> 2; kb = u & 3; Kdim = 128; Nout = 128; }
    else if (t < 80)   { W = W3; int u = t - 48;  fb = u >> 2; kb = u & 3; Kdim = 128; Nout = 128; }
    else if (t < 112)  { W = W4; int u = t - 80;  fb = u >> 2; kb = u & 3; Kdim = 128; Nout = 128; }
    else               { W = W5; fb = 0;          kb = t - 112;            Kdim = 128; Nout = 4;   }

    int n = fb * 16 + (lane & 15);                       // out feature
    int k = kb * 32 + ((lane >> 4) << 4) + h;
    float v = (k < Kdim && n < Nout) ? W[k * Nout + n] : 0.0f;   // W is [K][N] row-major
    ws[idx] = (_Float16)v;
}

// ---------------------------------------------------------------------------
// Main kernel helpers
// ---------------------------------------------------------------------------
union HU { v16h v; _Float16 h[16]; unsigned u[8]; };

__device__ __forceinline__ v16h loadA(const _Float16* __restrict__ w, int fragId, int lane) {
    return *(const v16h*)(w + (size_t)fragId * FRAG_HALVES + lane * 16);
}

// tanh-form GELU on a packed f16 pair: gelu(x) = 0.5x(1+tanh(0.79788456(x+0.044715x^3)))
// Preferred path uses the CDNA5 hardware v_tanh_f16 (1 trans/elem);
// fallback is sigmoid form via exp2+rcp (2 trans/elem).
__device__ __forceinline__ unsigned gelu2_pack(float x0, float x1) {
    fp16x2 pk = __builtin_amdgcn_cvt_pkrtz(x0, x1);      // v_cvt_pk_f16_f32 (rtz)
    h2 xh;
    __builtin_memcpy(&xh, &pk, 4);
    h2 x2 = xh * xh;
#if __has_builtin(__builtin_amdgcn_tanh_f16)
    // u = 0.79788456 x + 0.03567740814 x^3
    h2 u  = xh * ((_Float16)0.79788456f + x2 * (_Float16)0.03567741f);
    h2 t;
    t.x = (_Float16)__builtin_amdgcn_tanh_f16(u.x);
    t.y = (_Float16)__builtin_amdgcn_tanh_f16(u.y);
    h2 hx = xh * (_Float16)0.5f;
    h2 g  = hx * t + hx;                                  // 0.5x(1+tanh(u))
#else
    // arg = x * (-2.3022077 - 0.1029547 x^2)  [log2(e) folded], g = x/(1+exp2(arg))
    h2 a  = xh * ((_Float16)-2.3022077f + x2 * (_Float16)-0.1029547f);
    __half2 ah, th;
    __builtin_memcpy(&ah, &a, 4);
    th = h2exp2(ah);
    h2 t;
    __builtin_memcpy(&t, &th, 4);
    h2 d  = t + (_Float16)1.0f;
    __half2 dh, rh;
    __builtin_memcpy(&dh, &d, 4);
    rh = h2rcp(dh);
    h2 r;
    __builtin_memcpy(&r, &rh, 4);
    h2 g  = xh * r;
#endif
    unsigned uu;
    __builtin_memcpy(&uu, &g, 4);
    return uu;
}

// GELU + f16 pack + half-wave exchange -> B fragments for next layer.
// acc[fb] lane layout: col = sample = lane&15, row feature f = fb*16+r+halfId*8
// (bias already folded into the WMMA C operand).
// B frag kb needs per lane the 16 contiguous features FB*16..FB*16+15 of its
// sample, FB = 2*kb + halfId; low 8 live in the halfId==0 lane, high 8 in the
// halfId==1 lane -> shfl_xor(16) + select.
__device__ __forceinline__ void gelu_exchange(const v8f* acc, int halfId, HU* Bf) {
    unsigned own[8][4], oth[8][4];
#pragma unroll
    for (int fb = 0; fb < 8; ++fb) {
#pragma unroll
        for (int p = 0; p < 4; ++p)
            own[fb][p] = gelu2_pack(acc[fb][2 * p], acc[fb][2 * p + 1]);
    }
#pragma unroll
    for (int fb = 0; fb < 8; ++fb)
#pragma unroll
        for (int p = 0; p < 4; ++p)
            oth[fb][p] = (unsigned)__shfl_xor((int)own[fb][p], 16, 32);
#pragma unroll
    for (int kb = 0; kb < 4; ++kb) {
#pragma unroll
        for (int p = 0; p < 4; ++p) {
            Bf[kb].u[p]     = halfId ? oth[2 * kb + 1][p] : own[2 * kb][p];
            Bf[kb].u[4 + p] = halfId ? own[2 * kb + 1][p] : oth[2 * kb][p];
        }
    }
}

// bias broadcast into the C operand: C[r] = bias[fb*16 + halfId*8 + r]
__device__ __forceinline__ v8f bias_c(const float* __restrict__ bias, int fb, int halfId) {
    const float4* bp = (const float4*)(bias + fb * 16 + halfId * 8);
    float4 b0 = bp[0], b1 = bp[1];
    v8f c;
    c[0] = b0.x; c[1] = b0.y; c[2] = b0.z; c[3] = b0.w;
    c[4] = b1.x; c[5] = b1.y; c[6] = b1.z; c[7] = b1.w;
    return c;
}

__global__ __launch_bounds__(256, 2) void domino_main(
    const float* __restrict__ qp, const float* __restrict__ pts,
    const float* __restrict__ freqs,
    const float* __restrict__ b1, const float* __restrict__ b2,
    const float* __restrict__ b3, const float* __restrict__ b4,
    const float* __restrict__ b5,
    const int* __restrict__ mapping,
    const _Float16* __restrict__ wsA,
    float* __restrict__ out, int M) {
    const int lane   = threadIdx.x & 31;
    const int wave   = threadIdx.x >> 5;
    const int tile   = blockIdx.x * 8 + wave;           // 16 samples per tile
    const int halfId = lane >> 4;
    if (tile * 16 >= M * 8) return;

    // ---- gather + relative coords (sample = flat (m,k) index) -------------
    const int s = tile * 16 + (lane & 15);
    const int m = s >> 3;
    const int gi = mapping[s];
    float rel[3];
#pragma unroll
    for (int d = 0; d < 3; ++d)
        rel[d] = pts[gi * 3 + d] - qp[m * 3 + d];

    float fr[8];
#pragma unroll
    for (int f = 0; f < 8; ++f) fr[f] = freqs[f];

    // scaled[j] = rel[j%3] * freqs[j/3];  feat[f] = sin(sc[f]) f<24, cos(sc[f-24])
    float sc[24];
#pragma unroll
    for (int j = 0; j < 24; ++j)
        sc[j] = rel[j % 3] * fr[j / 3];

    // B fragments of F^T (48x16, K padded to 64): lane supplies features
    // kb*32 + halfId*16 + j of its sample.  sin(x) = cos(x - pi/2): select the
    // angle first, single cos per feature.
    const float PIH = 1.57079632679489662f;
    HU B0, B1;
#pragma unroll
    for (int j = 0; j < 16; ++j) {
        // B0: half0 -> f=j (sin);  half1 -> f=j+16 (sin if j<8 else cos)
        float a0 = sc[j] - PIH;
        float a1 = (j < 8) ? (sc[j + 16] - PIH) : sc[j - 8];
        float aa = halfId ? a1 : a0;
        B0.h[j] = (_Float16)__cosf(aa);
        // B1: half0 -> f=32+j (cos of sc[8+j]); half1 -> f=48+j (zero pad)
        float c1 = __cosf(sc[j + 8]);
        B1.h[j] = halfId ? (_Float16)0.0f : (_Float16)c1;
    }

    // ---- layer 1: T1 = W1^T (128x48) @ F^T, bias in C --------------------
    v8f acc[8];
#pragma unroll
    for (int fb = 0; fb < 8; ++fb) {
        v8f z = bias_c(b1, fb, halfId);
        z = __builtin_amdgcn_wmma_f32_16x16x32_f16(false, loadA(wsA, fb * 2 + 0, lane),
                                                   false, B0.v, (short)0, z, false, false);
        z = __builtin_amdgcn_wmma_f32_16x16x32_f16(false, loadA(wsA, fb * 2 + 1, lane),
                                                   false, B1.v, (short)0, z, false, false);
        acc[fb] = z;
    }

    HU Bf[4];
    gelu_exchange(acc, halfId, Bf);

    // ---- layers 2..4: 128x128, bias in C ---------------------------------
    const float* BIAS[3] = { b2, b3, b4 };
#pragma unroll
    for (int L = 0; L < 3; ++L) {
        const int base = 16 + L * 32;
        v8f nacc[8];
#pragma unroll
        for (int fb = 0; fb < 8; ++fb) {
            v8f z = bias_c(BIAS[L], fb, halfId);
#pragma unroll
            for (int kb = 0; kb < 4; ++kb)
                z = __builtin_amdgcn_wmma_f32_16x16x32_f16(
                        false, loadA(wsA, base + fb * 4 + kb, lane),
                        false, Bf[kb].v, (short)0, z, false, false);
            nacc[fb] = z;
        }
        gelu_exchange(nacc, halfId, Bf);
    }

    // ---- layer 5: 4 outputs (padded to 16 rows), bias applied post-mean --
    v8f z5 = {};
#pragma unroll
    for (int kb = 0; kb < 4; ++kb)
        z5 = __builtin_amdgcn_wmma_f32_16x16x32_f16(false, loadA(wsA, 112 + kb, lane),
                                                    false, Bf[kb].v, (short)0, z5, false, false);

    // ---- mean over K=8 neighbors + bias, vectorized store ----------------
    // z5: lane col = sample (lane&15), row r = out feature (halfId==0, r<4 valid)
    float red[4];
#pragma unroll
    for (int r = 0; r < 4; ++r) {
        float v = z5[r];
        v += __shfl_xor(v, 1, 32);
        v += __shfl_xor(v, 2, 32);
        v += __shfl_xor(v, 4, 32);
        red[r] = v;
    }
    if (halfId == 0 && (lane & 7) == 0) {
        const float4 bv = *(const float4*)b5;
        float4 o;
        o.x = red[0] * 0.125f + bv.x;
        o.y = red[1] * 0.125f + bv.y;
        o.z = red[2] * 0.125f + bv.z;
        o.w = red[3] * 0.125f + bv.w;
        *(float4*)(out + (s >> 3) * 4) = o;   // lane 0 -> m=tile*2, lane 8 -> m=tile*2+1
    }
}

// ---------------------------------------------------------------------------
extern "C" void kernel_launch(void* const* d_in, const int* in_sizes, int n_in,
                              void* d_out, int out_size, void* d_ws, size_t ws_size,
                              hipStream_t stream) {
    const float* qp    = (const float*)d_in[0];
    const float* pts   = (const float*)d_in[1];
    const float* freqs = (const float*)d_in[2];
    const float* W1    = (const float*)d_in[3];
    const float* b1    = (const float*)d_in[4];
    const float* W2    = (const float*)d_in[5];
    const float* b2    = (const float*)d_in[6];
    const float* W3    = (const float*)d_in[7];
    const float* b3    = (const float*)d_in[8];
    const float* W4    = (const float*)d_in[9];
    const float* b4    = (const float*)d_in[10];
    const float* W5    = (const float*)d_in[11];
    const float* b5    = (const float*)d_in[12];
    const int*   map   = (const int*)d_in[13];
    _Float16*    wsA   = (_Float16*)d_ws;
    float*       out   = (float*)d_out;

    const int M = in_sizes[0] / 3;                 // 131072

    // 1) swizzle weights into f16 A-fragments in workspace
    domino_prep_weights<<<(WS_HALVES + 255) / 256, 256, 0, stream>>>(
        W1, W2, W3, W4, W5, wsA, WS_HALVES);

    // 2) fused gather + fourier + 5-layer WMMA MLP + K-mean
    const int blocks = (M * 8) / 128;              // 8 waves x 16 samples per block
    domino_main<<<blocks, 256, 0, stream>>>(
        qp, pts, freqs, b1, b2, b3, b4, b5, map, wsA, out, M);
}
